// SpMiddleResNetFHD_20796231647882
// MI455X (gfx1250) — matrix operand order repaired and checked by reading, not compiled, not run on hardware
//
#include <hip/hip_runtime.h>
#include <hip/hip_bf16.h>

// ---------------------------------------------------------------------------
// SpMiddleResNetFHD on MI455X (gfx1250, wave32, WMMA).
//
//  * Dense-masked sparse conv == implicit GEMM:
//      D[16 cout x 16 voxels] += A[16 cout x 32 K] * B[32 K x 16 voxels]
//    via V_WMMA_F32_16X16X32_BF16 (f32 accumulate).
//  * Ci>=32 levels: K = 32 input channels per tap (generic kernel, NC chunks).
//  * Ci==16 levels (stem/L0/d1 -- the dominant cost): K = TWO TAPS x 16 ch,
//    so no zero-padding waste: 14 WMMAs per tile instead of 27.
//  * Activations channel-last bf16 -> B fragment = ONE aligned 32B load/lane.
//  * Weights pre-packed in WMMA A-FRAGMENT ORDER -> ONE aligned 32B load/lane.
//  * BN folded to scale/shift; BN+ReLU+residual+mask fused in epilogue.
//  * Sparsity: wave-uniform __ballot over the 16 output voxels' mask skips
//    the whole K loop for inactive tiles (EXEC all-ones around WMMA).
// ---------------------------------------------------------------------------

typedef __bf16 bf16_t;
typedef __attribute__((ext_vector_type(16))) bf16_t v16bf;
typedef __attribute__((ext_vector_type(8)))  bf16_t v8bf;
typedef __attribute__((ext_vector_type(8)))  float  v8f;
typedef __attribute__((ext_vector_type(4)))  float  v4f;

#define WAVES_PER_BLOCK 8

// ---------------- helper kernels ----------------

__global__ void k_scatter(const float* __restrict__ feat,
                          const int* __restrict__ coors,
                          bf16_t* __restrict__ dense,
                          unsigned char* __restrict__ mask,
                          int N, int H, int W, int Cs) {
  int n = blockIdx.x * blockDim.x + threadIdx.x;
  if (n >= N) return;
  int d = coors[n * 4 + 1];
  int h = coors[n * 4 + 2];
  int w = coors[n * 4 + 3];
  unsigned v = (unsigned)((d * H + h) * W + w);
  mask[v] = 1;
  bf16_t* p = dense + (size_t)v * (unsigned)Cs;
  #pragma unroll
  for (int c = 0; c < 5; ++c) p[c] = (bf16_t)feat[n * 5 + c];
}

__global__ void k_bnprep(const float* __restrict__ g, const float* __restrict__ b,
                         const float* __restrict__ m, const float* __restrict__ v,
                         float* __restrict__ scale, float* __restrict__ shift, int C) {
  int c = blockIdx.x * blockDim.x + threadIdx.x;
  if (c >= C) return;
  float s = g[c] * rsqrtf(v[c] + 1e-3f);
  scale[c] = s;
  shift[c] = b[c] - m[c] * s;
}

// Generic packer: src f32 [Co][Ci][KO] -> bf16 A-fragment order
// [o][kc][cotile][lane][e], K = 32 input channels per tap (zero pad >= Ci).
__global__ void k_wfrag(const float* __restrict__ src, bf16_t* __restrict__ dst,
                        int Co, int Ci, int nChunks, int KO) {
  int nCoT = Co >> 4;
  unsigned total = (unsigned)(KO * nChunks * nCoT * 512);
  unsigned i = blockIdx.x * blockDim.x + threadIdx.x;
  if (i >= total) return;
  int e    = (int)(i & 15u);
  int lane = (int)((i >> 4) & 31u);
  unsigned t = i >> 9;
  int cy   = (int)(t % (unsigned)nCoT);
  t /= (unsigned)nCoT;
  int kc   = (int)(t % (unsigned)nChunks);
  int o    = (int)(t / (unsigned)nChunks);
  int m = lane & 15;
  int khalf = (lane >> 4) * 16;
  int k = khalf + e;
  int ci = kc * 32 + k;
  int co = cy * 16 + m;
  float val = (ci < Ci) ? src[(size_t)(unsigned)(co * Ci + ci) * (unsigned)KO + o] : 0.0f;
  dst[i] = (bf16_t)val;
}

// Tap-pair packer for Ci<=16, 3x3x3: K = 2 taps x 16 channels.
// dst: [14 pairs][cotile][lane][e]; K -> tap = op*2 + (K>>4), ci = K&15.
__global__ void k_wfrag_pair(const float* __restrict__ src, bf16_t* __restrict__ dst,
                             int Co, int Ci) {
  int nCoT = Co >> 4;
  unsigned total = (unsigned)(14 * nCoT * 512);
  unsigned i = blockIdx.x * blockDim.x + threadIdx.x;
  if (i >= total) return;
  int e    = (int)(i & 15u);
  int lane = (int)((i >> 4) & 31u);
  unsigned t = i >> 9;
  int cy = (int)(t % (unsigned)nCoT);
  int op = (int)(t / (unsigned)nCoT);
  int m = lane & 15;
  int khalf = (lane >> 4) * 16;
  int K = khalf + e;
  int tap = op * 2 + (K >> 4);
  int ci = K & 15;
  int co = cy * 16 + m;
  float val = (tap < 27 && ci < Ci)
                  ? src[(size_t)(unsigned)(co * Ci + ci) * 27u + tap] : 0.0f;
  dst[i] = (bf16_t)val;
}

__global__ void k_downmask(const unsigned char* __restrict__ mi,
                           unsigned char* __restrict__ mo,
                           int Di, int Hi, int Wi, int Do, int Ho, int Wo,
                           int KD, int KH, int KW,
                           int sd, int sh, int sw, int pd, int ph, int pw) {
  unsigned v = blockIdx.x * blockDim.x + threadIdx.x;
  unsigned DHWo = (unsigned)(Do * Ho * Wo);
  if (v >= DHWo) return;
  int ow = (int)(v % (unsigned)Wo);
  unsigned t = v / (unsigned)Wo;
  int oh = (int)(t % (unsigned)Ho);
  int od = (int)(t / (unsigned)Ho);
  unsigned char r = 0;
  for (int kd = 0; kd < KD; ++kd) {
    int id = od * sd - pd + kd;
    if ((unsigned)id >= (unsigned)Di) continue;
    for (int kh = 0; kh < KH; ++kh) {
      int ih = oh * sh - ph + kh;
      if ((unsigned)ih >= (unsigned)Hi) continue;
      for (int kw = 0; kw < KW; ++kw) {
        int iw = ow * sw - pw + kw;
        if ((unsigned)iw >= (unsigned)Wi) continue;
        if (mi[(unsigned)((id * Hi + ih) * Wi + iw)]) r = 1;
      }
    }
  }
  mo[v] = r;
}

// ---------------- generic implicit-GEMM WMMA conv (Ci >= 32) ----------------

template <int NC>
__global__ __launch_bounds__(256) void k_conv_wmma(
    const bf16_t* __restrict__ in, const bf16_t* __restrict__ wt,
    const float* __restrict__ bnscale, const float* __restrict__ bnshift,
    const unsigned char* __restrict__ mask,
    const bf16_t* __restrict__ res,
    bf16_t* __restrict__ out, float* __restrict__ outf,
    int Di, int Hi, int Wi, int Do, int Ho, int Wo,
    int Cs, int Co,
    int KD, int KH, int KW, int sd, int sh, int sw, int pd, int ph, int pw,
    int do_relu) {
  const int lane = threadIdx.x & 31;
  const int wave = threadIdx.x >> 5;
  const int n  = lane & 15;
  const int hi = lane >> 4;
  const unsigned DHWo = (unsigned)(Do * Ho * Wo);
  const unsigned tile = blockIdx.x * WAVES_PER_BLOCK + wave;
  const unsigned v = tile * 16u + (unsigned)n;
  const bool vok = v < DHWo;
  int od = 0, oh = 0, ow = 0;
  if (vok) {
    ow = (int)(v % (unsigned)Wo);
    unsigned t = v / (unsigned)Wo;
    oh = (int)(t % (unsigned)Ho);
    od = (int)(t / (unsigned)Ho);
  }
  const unsigned char mk = vok ? mask[v] : (unsigned char)0;

  v8f acc = {};
  const unsigned long long active = __ballot(mk != 0);
  const int nCoT = Co >> 4;
  const int cotile = blockIdx.y * 16;

  if (active) {
    const int wStepChunk = nCoT * 512;
    const bf16_t* __restrict__ wl = wt + (blockIdx.y * 32 + lane) * 16;
    const int HWi = Hi * Wi;
    const int iw0 = ow * sw - pw;
    int o = 0;
    for (int kd = 0; kd < KD; ++kd) {
      const int id = od * sd - pd + kd;
      const bool dok = vok & ((unsigned)id < (unsigned)Di);
      for (int kh = 0; kh < KH; ++kh) {
        const int ih = oh * sh - ph + kh;
        const bool hok = dok & ((unsigned)ih < (unsigned)Hi);
        const int rowBase = id * HWi + ih * Wi;
        for (int kw = 0; kw < KW; ++kw, ++o) {
          const int iw = iw0 + kw;
          const bool iok = hok & ((unsigned)iw < (unsigned)Wi);
          const bf16_t* __restrict__ ib =
              in + (size_t)(unsigned)(rowBase + iw) * (unsigned)Cs;
          if (iok) __builtin_prefetch(ib + Cs, 0, 1);
          const bf16_t* __restrict__ wo = wl + (size_t)(unsigned)(o * NC) * wStepChunk;
          #pragma unroll
          for (int kc = 0; kc < NC; ++kc) {
            const int cb = kc * 32 + hi * 16;
            v16bf bfrag = {};
            if (iok && cb < Cs) bfrag = *(const v16bf*)(ib + cb);
            const v16bf afrag = *(const v16bf*)(wo + (size_t)(kc * wStepChunk));
            acc = __builtin_amdgcn_wmma_f32_16x16x32_bf16(
                false, afrag, false, bfrag, (short)0, acc, false, false);
          }
        }
      }
    }
  }

  if (!vok) return;
  const int cobase = cotile + hi * 8;
  const v4f s0 = *(const v4f*)(bnscale + cobase);
  const v4f s1 = *(const v4f*)(bnscale + cobase + 4);
  const v4f h0 = *(const v4f*)(bnshift + cobase);
  const v4f h1 = *(const v4f*)(bnshift + cobase + 4);
  if (out) {
    v8bf ob;
    if (res) {
      const v8bf rv = *(const v8bf*)(res + (size_t)v * (unsigned)Co + cobase);
      #pragma unroll
      for (int r = 0; r < 8; ++r) {
        const float sc = (r < 4) ? s0[r & 3] : s1[r & 3];
        const float sh_ = (r < 4) ? h0[r & 3] : h1[r & 3];
        float y = acc[r] * sc + sh_ + (float)rv[r];
        if (do_relu) y = fmaxf(y, 0.0f);
        ob[r] = mk ? (bf16_t)y : (bf16_t)0.0f;
      }
    } else {
      #pragma unroll
      for (int r = 0; r < 8; ++r) {
        const float sc = (r < 4) ? s0[r & 3] : s1[r & 3];
        const float sh_ = (r < 4) ? h0[r & 3] : h1[r & 3];
        float y = acc[r] * sc + sh_;
        if (do_relu) y = fmaxf(y, 0.0f);
        ob[r] = mk ? (bf16_t)y : (bf16_t)0.0f;
      }
    }
    *(v8bf*)(out + (size_t)v * (unsigned)Co + cobase) = ob;
  } else {
    #pragma unroll
    for (int r = 0; r < 8; ++r) {
      const float sc = (r < 4) ? s0[r & 3] : s1[r & 3];
      const float sh_ = (r < 4) ? h0[r & 3] : h1[r & 3];
      float y = acc[r] * sc + sh_;
      if (do_relu) y = fmaxf(y, 0.0f);
      outf[(size_t)(unsigned)(cobase + r) * DHWo + v] = mk ? y : 0.0f;
    }
  }
}

// ------------- tap-pair WMMA conv: Cs == 16, 3x3x3 kernel -------------------
// K dimension = 2 taps x 16 input channels: lane half hi gathers the voxel of
// tap (o+hi); fully unrolled 14 pairs (tap 27 is a zero pad).

__global__ __launch_bounds__(256) void k_conv_wmma_c16(
    const bf16_t* __restrict__ in,        // [DHWi][16]
    const bf16_t* __restrict__ wt,        // [14][CoT][32][16] pair-frag order
    const float* __restrict__ bnscale, const float* __restrict__ bnshift,
    const unsigned char* __restrict__ mask,
    const bf16_t* __restrict__ res,
    bf16_t* __restrict__ out,             // [DHWo][Co]
    int Di, int Hi, int Wi, int Do, int Ho, int Wo, int Co,
    int sd, int sh, int sw, int pd, int ph, int pw) {
  const int lane = threadIdx.x & 31;
  const int wave = threadIdx.x >> 5;
  const int n  = lane & 15;
  const int hi = lane >> 4;
  const unsigned DHWo = (unsigned)(Do * Ho * Wo);
  const unsigned tile = blockIdx.x * WAVES_PER_BLOCK + wave;
  const unsigned v = tile * 16u + (unsigned)n;
  const bool vok = v < DHWo;
  int od = 0, oh = 0, ow = 0;
  if (vok) {
    ow = (int)(v % (unsigned)Wo);
    unsigned t = v / (unsigned)Wo;
    oh = (int)(t % (unsigned)Ho);
    od = (int)(t / (unsigned)Ho);
  }
  const unsigned char mk = vok ? mask[v] : (unsigned char)0;

  v8f acc = {};
  const unsigned long long active = __ballot(mk != 0);
  const int cotile = blockIdx.y * 16;

  if (active) {
    const int wStep = (Co >> 4) * 512;
    const bf16_t* __restrict__ wl = wt + (blockIdx.y * 32 + lane) * 16;
    const int HWi = Hi * Wi;
    const int id0 = od * sd - pd;
    const int ih0 = oh * sh - ph;
    const int iw0 = ow * sw - pw;
    #pragma unroll
    for (int o = 0; o < 28; o += 2) {
      const int tap = o + hi;            // this lane-half's tap
      const int kd = tap / 9;
      const int r9 = tap - kd * 9;
      const int kh = r9 / 3;
      const int kw = r9 - kh * 3;
      const int id = id0 + kd;
      const int ih = ih0 + kh;
      const int iw = iw0 + kw;
      const bool iok = vok & (tap < 27) &
                       ((unsigned)id < (unsigned)Di) &
                       ((unsigned)ih < (unsigned)Hi) &
                       ((unsigned)iw < (unsigned)Wi);
      v16bf bfrag = {};
      if (iok)
        bfrag = *(const v16bf*)(in + (size_t)(unsigned)(id * HWi + ih * Wi + iw) * 16u);
      const v16bf afrag = *(const v16bf*)(wl + (size_t)(unsigned)((o >> 1) * wStep));
      acc = __builtin_amdgcn_wmma_f32_16x16x32_bf16(
          false, afrag, false, bfrag, (short)0, acc, false, false);
    }
  }

  if (!vok) return;
  const int cobase = cotile + hi * 8;
  const v4f s0 = *(const v4f*)(bnscale + cobase);
  const v4f s1 = *(const v4f*)(bnscale + cobase + 4);
  const v4f h0 = *(const v4f*)(bnshift + cobase);
  const v4f h1 = *(const v4f*)(bnshift + cobase + 4);
  v8bf ob;
  if (res) {
    const v8bf rv = *(const v8bf*)(res + (size_t)v * (unsigned)Co + cobase);
    #pragma unroll
    for (int r = 0; r < 8; ++r) {
      const float sc = (r < 4) ? s0[r & 3] : s1[r & 3];
      const float sh_ = (r < 4) ? h0[r & 3] : h1[r & 3];
      float y = fmaxf(acc[r] * sc + sh_ + (float)rv[r], 0.0f);
      ob[r] = mk ? (bf16_t)y : (bf16_t)0.0f;
    }
  } else {
    #pragma unroll
    for (int r = 0; r < 8; ++r) {
      const float sc = (r < 4) ? s0[r & 3] : s1[r & 3];
      const float sh_ = (r < 4) ? h0[r & 3] : h1[r & 3];
      float y = fmaxf(acc[r] * sc + sh_, 0.0f);
      ob[r] = mk ? (bf16_t)y : (bf16_t)0.0f;
    }
  }
  *(v8bf*)(out + (size_t)v * (unsigned)Co + cobase) = ob;
}

// ---------------- host side ----------------

static inline size_t align_up(size_t x, size_t a) { return (x + a - 1) & ~(a - 1); }

struct CvCfg { int w, bn, Ci, Co, KD, KH, KW; };

static void launch_conv(hipStream_t s, const bf16_t* in, const bf16_t* wt,
                        const float* sc, const float* sf, const unsigned char* mk,
                        const bf16_t* res, bf16_t* out, float* outf,
                        int Di, int Hi, int Wi, int Do, int Ho, int Wo,
                        int Cs, int Co, int nC,
                        int KD, int KH, int KW, int sd, int sh, int sw,
                        int pd, int ph, int pw) {
  long DHWo = (long)Do * Ho * Wo;
  int tiles = (int)((DHWo + 15) / 16);
  dim3 grid((tiles + WAVES_PER_BLOCK - 1) / WAVES_PER_BLOCK, Co / 16);
  switch (nC) {
    case 1:
      k_conv_wmma<1><<<grid, 256, 0, s>>>(in, wt, sc, sf, mk, res, out, outf,
                                          Di, Hi, Wi, Do, Ho, Wo, Cs, Co,
                                          KD, KH, KW, sd, sh, sw, pd, ph, pw, 1);
      break;
    case 2:
      k_conv_wmma<2><<<grid, 256, 0, s>>>(in, wt, sc, sf, mk, res, out, outf,
                                          Di, Hi, Wi, Do, Ho, Wo, Cs, Co,
                                          KD, KH, KW, sd, sh, sw, pd, ph, pw, 1);
      break;
    default:
      k_conv_wmma<4><<<grid, 256, 0, s>>>(in, wt, sc, sf, mk, res, out, outf,
                                          Di, Hi, Wi, Do, Ho, Wo, Cs, Co,
                                          KD, KH, KW, sd, sh, sw, pd, ph, pw, 1);
      break;
  }
}

static void launch_conv16(hipStream_t s, const bf16_t* in, const bf16_t* wt,
                          const float* sc, const float* sf, const unsigned char* mk,
                          const bf16_t* res, bf16_t* out,
                          int Di, int Hi, int Wi, int Do, int Ho, int Wo, int Co,
                          int sd, int sh, int sw, int pd, int ph, int pw) {
  long DHWo = (long)Do * Ho * Wo;
  int tiles = (int)((DHWo + 15) / 16);
  dim3 grid((tiles + WAVES_PER_BLOCK - 1) / WAVES_PER_BLOCK, Co / 16);
  k_conv_wmma_c16<<<grid, 256, 0, s>>>(in, wt, sc, sf, mk, res, out,
                                       Di, Hi, Wi, Do, Ho, Wo, Co,
                                       sd, sh, sw, pd, ph, pw);
}

static void launch_downmask(hipStream_t s, const unsigned char* mi, unsigned char* mo,
                            int Di, int Hi, int Wi, int Do, int Ho, int Wo,
                            int KD, int KH, int KW, int sd, int sh, int sw,
                            int pd, int ph, int pw) {
  long DHWo = (long)Do * Ho * Wo;
  int blocks = (int)((DHWo + 255) / 256);
  k_downmask<<<blocks, 256, 0, s>>>(mi, mo, Di, Hi, Wi, Do, Ho, Wo,
                                    KD, KH, KW, sd, sh, sw, pd, ph, pw);
}

extern "C" void kernel_launch(void* const* d_in, const int* in_sizes, int n_in,
                              void* d_out, int out_size, void* d_ws, size_t ws_size,
                              hipStream_t stream) {
  (void)in_sizes; (void)n_in; (void)out_size; (void)ws_size;
  // d_in flattening (top-level insertion order, nested dicts by sorted key):
  // 0: voxel_features; 1..105: params leaves; 106: coors; 107: batch_size; 108: input_shape
  const float* feat = (const float*)d_in[0];
  const int* coors = (const int*)d_in[106];
  const int N = in_sizes[0] / 5;

  const int D0 = 41, H0 = 200, W0 = 200;
  const int D1 = 21, H1 = 100, W1 = 100;
  const int D2 = 11, H2 = 50,  W2 = 50;
  const int D3 = 5,  H3 = 25,  W3 = 25;
  const int D4 = 2,  H4 = 25,  W4 = 25;
  const long V0 = (long)D0 * H0 * W0, V1 = (long)D1 * H1 * W1;
  const long V2 = (long)D2 * H2 * W2, V3 = (long)D3 * H3 * W3, V4 = (long)D4 * H4 * W4;

  const CvCfg cfg[21] = {
    {105, 101,   5,  16, 3, 3, 3},                                 // 0 stem      (pair)
    {  9,   1,  16,  16, 3, 3, 3}, { 10,   5,  16,  16, 3, 3, 3},  // b0a         (pair)
    { 19,  11,  16,  16, 3, 3, 3}, { 20,  15,  16,  16, 3, 3, 3},  // b0b         (pair)
    { 85,  81,  16,  32, 3, 3, 3},                                 // 5 d1        (pair)
    { 29,  21,  32,  32, 3, 3, 3}, { 30,  25,  32,  32, 3, 3, 3},  // b1a
    { 39,  31,  32,  32, 3, 3, 3}, { 40,  35,  32,  32, 3, 3, 3},  // b1b
    { 90,  86,  32,  64, 3, 3, 3},                                 // 10 d2
    { 49,  41,  64,  64, 3, 3, 3}, { 50,  45,  64,  64, 3, 3, 3},  // b2a
    { 59,  51,  64,  64, 3, 3, 3}, { 60,  55,  64,  64, 3, 3, 3},  // b2b
    { 95,  91,  64, 128, 3, 3, 3},                                 // 15 d3
    { 69,  61, 128, 128, 3, 3, 3}, { 70,  65, 128, 128, 3, 3, 3},  // b3a
    { 79,  71, 128, 128, 3, 3, 3}, { 80,  75, 128, 128, 3, 3, 3},  // b3b
    {100,  96, 128, 128, 3, 1, 1},                                 // 20 d4
  };

  // ---- bump allocator over d_ws ----
  char* base = (char*)d_ws;
  size_t off = 0;
  auto alloc = [&](size_t bytes) -> char* {
    char* p = base + off;
    off = align_up(off + bytes, 256);
    return p;
  };
  bf16_t* x0[3]; for (int i = 0; i < 3; ++i) x0[i] = (bf16_t*)alloc(V0 * 16 * 2);
  bf16_t* x1[3]; for (int i = 0; i < 3; ++i) x1[i] = (bf16_t*)alloc(V1 * 32 * 2);
  bf16_t* x2[3]; for (int i = 0; i < 3; ++i) x2[i] = (bf16_t*)alloc(V2 * 64 * 2);
  bf16_t* x3[3]; for (int i = 0; i < 3; ++i) x3[i] = (bf16_t*)alloc(V3 * 128 * 2);
  unsigned char* m0 = (unsigned char*)alloc(V0);
  unsigned char* m1 = (unsigned char*)alloc(V1);
  unsigned char* m2 = (unsigned char*)alloc(V2);
  unsigned char* m3 = (unsigned char*)alloc(V3);
  unsigned char* m4 = (unsigned char*)alloc(V4);

  bf16_t* wt[21]; float* sc[21]; float* sf[21]; int nck[21];
  for (int i = 0; i < 21; ++i) {
    int KO = cfg[i].KD * cfg[i].KH * cfg[i].KW;
    nck[i] = (cfg[i].Ci + 31) / 32;
    bool pair = (i <= 5);  // Ci<=16, 3x3x3 levels use tap-pair packing
    size_t elems = pair ? (size_t)14 * (cfg[i].Co >> 4) * 512
                        : (size_t)KO * nck[i] * 32 * cfg[i].Co;
    wt[i] = (bf16_t*)alloc(elems * 2);
    sc[i] = (float*)alloc(cfg[i].Co * 4);
    sf[i] = (float*)alloc(cfg[i].Co * 4);
  }

  // ---- weight / BN prep (deterministic every call) ----
  for (int i = 0; i < 21; ++i) {
    int KO = cfg[i].KD * cfg[i].KH * cfg[i].KW;
    int Co = cfg[i].Co;
    const float* bnb = (const float*)d_in[cfg[i].bn + 0];
    const float* bng = (const float*)d_in[cfg[i].bn + 1];
    const float* bnm = (const float*)d_in[cfg[i].bn + 2];
    const float* bnv = (const float*)d_in[cfg[i].bn + 3];
    k_bnprep<<<(Co + 255) / 256, 256, 0, stream>>>(bng, bnb, bnm, bnv, sc[i], sf[i], Co);
    if (i <= 5) {
      long total = (long)14 * (Co >> 4) * 512;
      k_wfrag_pair<<<(int)((total + 255) / 256), 256, 0, stream>>>(
          (const float*)d_in[cfg[i].w], wt[i], Co, cfg[i].Ci);
    } else {
      long total = (long)KO * nck[i] * (Co >> 4) * 512;
      k_wfrag<<<(int)((total + 255) / 256), 256, 0, stream>>>(
          (const float*)d_in[cfg[i].w], wt[i], Co, cfg[i].Ci, nck[i], KO);
    }
  }

  // ---- scatter voxels (stem input stored as 16 ch, 5 real + 11 zero) ----
  hipMemsetAsync(x0[0], 0, (size_t)V0 * 16 * 2, stream);
  hipMemsetAsync(m0, 0, (size_t)V0, stream);
  k_scatter<<<(N + 255) / 256, 256, 0, stream>>>(feat, coors, x0[0], m0, N, H0, W0, 16);

  // ---- stem + level-0 blocks (tap-pair kernel) ----
  launch_conv16(stream, x0[0], wt[0], sc[0], sf[0], m0, nullptr, x0[1],
                D0, H0, W0, D0, H0, W0, 16, 1, 1, 1, 1, 1, 1);
  launch_conv16(stream, x0[1], wt[1], sc[1], sf[1], m0, nullptr, x0[2],
                D0, H0, W0, D0, H0, W0, 16, 1, 1, 1, 1, 1, 1);
  launch_conv16(stream, x0[2], wt[2], sc[2], sf[2], m0, x0[1], x0[0],
                D0, H0, W0, D0, H0, W0, 16, 1, 1, 1, 1, 1, 1);
  launch_conv16(stream, x0[0], wt[3], sc[3], sf[3], m0, nullptr, x0[1],
                D0, H0, W0, D0, H0, W0, 16, 1, 1, 1, 1, 1, 1);
  launch_conv16(stream, x0[1], wt[4], sc[4], sf[4], m0, x0[0], x0[2],
                D0, H0, W0, D0, H0, W0, 16, 1, 1, 1, 1, 1, 1);

  // ---- down1 (tap-pair, stride 2) + level-1 blocks ----
  launch_downmask(stream, m0, m1, D0, H0, W0, D1, H1, W1, 3, 3, 3, 2, 2, 2, 1, 1, 1);
  launch_conv16(stream, x0[2], wt[5], sc[5], sf[5], m1, nullptr, x1[0],
                D0, H0, W0, D1, H1, W1, 32, 2, 2, 2, 1, 1, 1);
  launch_conv(stream, x1[0], wt[6], sc[6], sf[6], m1, nullptr, x1[1], nullptr,
              D1, H1, W1, D1, H1, W1, 32, 32, 1, 3, 3, 3, 1, 1, 1, 1, 1, 1);
  launch_conv(stream, x1[1], wt[7], sc[7], sf[7], m1, x1[0], x1[2], nullptr,
              D1, H1, W1, D1, H1, W1, 32, 32, 1, 3, 3, 3, 1, 1, 1, 1, 1, 1);
  launch_conv(stream, x1[2], wt[8], sc[8], sf[8], m1, nullptr, x1[0], nullptr,
              D1, H1, W1, D1, H1, W1, 32, 32, 1, 3, 3, 3, 1, 1, 1, 1, 1, 1);
  launch_conv(stream, x1[0], wt[9], sc[9], sf[9], m1, x1[2], x1[1], nullptr,
              D1, H1, W1, D1, H1, W1, 32, 32, 1, 3, 3, 3, 1, 1, 1, 1, 1, 1);

  // ---- down2 + level-2 blocks ----
  launch_downmask(stream, m1, m2, D1, H1, W1, D2, H2, W2, 3, 3, 3, 2, 2, 2, 1, 1, 1);
  launch_conv(stream, x1[1], wt[10], sc[10], sf[10], m2, nullptr, x2[0], nullptr,
              D1, H1, W1, D2, H2, W2, 32, 64, 1, 3, 3, 3, 2, 2, 2, 1, 1, 1);
  launch_conv(stream, x2[0], wt[11], sc[11], sf[11], m2, nullptr, x2[1], nullptr,
              D2, H2, W2, D2, H2, W2, 64, 64, 2, 3, 3, 3, 1, 1, 1, 1, 1, 1);
  launch_conv(stream, x2[1], wt[12], sc[12], sf[12], m2, x2[0], x2[2], nullptr,
              D2, H2, W2, D2, H2, W2, 64, 64, 2, 3, 3, 3, 1, 1, 1, 1, 1, 1);
  launch_conv(stream, x2[2], wt[13], sc[13], sf[13], m2, nullptr, x2[0], nullptr,
              D2, H2, W2, D2, H2, W2, 64, 64, 2, 3, 3, 3, 1, 1, 1, 1, 1, 1);
  launch_conv(stream, x2[0], wt[14], sc[14], sf[14], m2, x2[2], x2[1], nullptr,
              D2, H2, W2, D2, H2, W2, 64, 64, 2, 3, 3, 3, 1, 1, 1, 1, 1, 1);

  // ---- down3 (pad d=0, h=w=1) + level-3 blocks ----
  launch_downmask(stream, m2, m3, D2, H2, W2, D3, H3, W3, 3, 3, 3, 2, 2, 2, 0, 1, 1);
  launch_conv(stream, x2[1], wt[15], sc[15], sf[15], m3, nullptr, x3[0], nullptr,
              D2, H2, W2, D3, H3, W3, 64, 128, 2, 3, 3, 3, 2, 2, 2, 0, 1, 1);
  launch_conv(stream, x3[0], wt[16], sc[16], sf[16], m3, nullptr, x3[1], nullptr,
              D3, H3, W3, D3, H3, W3, 128, 128, 4, 3, 3, 3, 1, 1, 1, 1, 1, 1);
  launch_conv(stream, x3[1], wt[17], sc[17], sf[17], m3, x3[0], x3[2], nullptr,
              D3, H3, W3, D3, H3, W3, 128, 128, 4, 3, 3, 3, 1, 1, 1, 1, 1, 1);
  launch_conv(stream, x3[2], wt[18], sc[18], sf[18], m3, nullptr, x3[0], nullptr,
              D3, H3, W3, D3, H3, W3, 128, 128, 4, 3, 3, 3, 1, 1, 1, 1, 1, 1);
  launch_conv(stream, x3[0], wt[19], sc[19], sf[19], m3, x3[2], x3[1], nullptr,
              D3, H3, W3, D3, H3, W3, 128, 128, 4, 3, 3, 3, 1, 1, 1, 1, 1, 1);

  // ---- down4: (3,1,1) stride (2,1,1) pad 0 -> f32 straight into d_out ----
  launch_downmask(stream, m3, m4, D3, H3, W3, D4, H4, W4, 3, 1, 1, 2, 1, 1, 0, 0, 0);
  launch_conv(stream, x3[1], wt[20], sc[20], sf[20], m4, nullptr, nullptr, (float*)d_out,
              D3, H3, W3, D4, H4, W4, 128, 128, 4, 3, 1, 1, 2, 1, 1, 0, 0, 0);
}
